// HungarianAssigner_8177617731832
// MI455X (gfx1250) — compile-verified, matrix operand order checked by reading, not run on hardware
//
#include <hip/hip_runtime.h>

// ---------------------------------------------------------------------------
// Hungarian matcher cost matrix for MI455X (gfx1250, wave32, WMMA).
//
//   out[b,n,m] = cost_class + cost_mask + cost_dice
// with S = sum_l p*t computed by v_wmma_f32_16x16x32_f16, and sum_p / sum_t
// obtained for free via an all-ones B column (m==20) and an all-ones A row
// (padded n==111). Inner loop is branchless and software-pipelined (ping-pong
// load tiles); sigmoid uses v_exp_f32 + v_rcp_f32 (result is truncated to f16
// anyway, so the 1-ulp hardware rcp is plenty).
// ---------------------------------------------------------------------------

typedef __attribute__((ext_vector_type(16))) _Float16 v16h;
typedef __attribute__((ext_vector_type(8)))  float    v8f;

#define BB 4
#define NN 100
#define CC 81
#define MM 20
#define LL 65536

#define NT 7            // ceil(100/16) n-tiles
#define NP (NT * 16)    // 112 padded rows
#define MP 32           // padded cols (2 m-tiles)

#define LSPLIT 128
#define LCHUNK (LL / LSPLIT)   // 512
#define KITERS (LCHUNK / 32)   // 16 K=32 steps per wave (even)

__device__ __forceinline__ float sigmoidf(float x) {
    // v_exp_f32 + v_add_f32 + v_rcp_f32 (no IEEE div sequence)
    return __builtin_amdgcn_rcpf(1.0f + __expf(-x));
}

// One K=32 step worth of raw f32 data: 4 float4 each for A, B0, B1 streams.
struct KTile {
    float4 x[12];
};

__device__ __forceinline__ void load_tiles(KTile& t,
                                           const float* __restrict__ aRow,
                                           const float* __restrict__ tRow0,
                                           const float* __restrict__ tRow1,
                                           int lbase, int o1, int o2) {
    const float4* pa  = (const float4*)(aRow  + lbase);
    const float4* pt0 = (const float4*)(tRow0 + lbase);
    const float4* pt1 = (const float4*)(tRow1 + lbase);
    t.x[0]  = pa[o1];  t.x[1]  = pa[o1 + 1];  t.x[2]  = pa[o2];  t.x[3]  = pa[o2 + 1];
    t.x[4]  = pt0[o1]; t.x[5]  = pt0[o1 + 1]; t.x[6]  = pt0[o2]; t.x[7]  = pt0[o2 + 1];
    t.x[8]  = pt1[o1]; t.x[9]  = pt1[o1 + 1]; t.x[10] = pt1[o2]; t.x[11] = pt1[o2 + 1];
}

__device__ __forceinline__ void process_tile(const KTile& t,
                                             float aMul, float aAdd,
                                             float bMul, float bAdd,
                                             v8f& c0, v8f& c1) {
    const float* f = (const float*)t.x;
    v16h a, bt0, bt1;
    #pragma unroll
    for (int i = 0; i < 16; ++i)
        a[i] = (_Float16)__builtin_fmaf(sigmoidf(f[i]), aMul, aAdd);
    #pragma unroll
    for (int i = 0; i < 16; ++i)
        bt0[i] = (_Float16)f[16 + i];
    #pragma unroll
    for (int i = 0; i < 16; ++i)
        bt1[i] = (_Float16)__builtin_fmaf(f[32 + i], bMul, bAdd);

    c0 = __builtin_amdgcn_wmma_f32_16x16x32_f16(false, a, false, bt0,
                                                (short)0, c0, false, false);
    c1 = __builtin_amdgcn_wmma_f32_16x16x32_f16(false, a, false, bt1,
                                                (short)0, c1, false, false);
}

// ---------------------------------------------------------------------------
// Zero the accumulator workspace (deterministic each launch).
// ---------------------------------------------------------------------------
__global__ void hung_zero_ws(float* __restrict__ ws) {
    int i = blockIdx.x * blockDim.x + threadIdx.x;
    if (i < BB * NP * MP) ws[i] = 0.0f;
}

// ---------------------------------------------------------------------------
// Main WMMA kernel: grid = (LSPLIT, NT, B), block = one wave32.
// ---------------------------------------------------------------------------
__global__ __launch_bounds__(32)
void hung_cost_wmma(const float* __restrict__ pred_masks,
                    const float* __restrict__ gt_masks,
                    float* __restrict__ ws) {
    const int lane = threadIdx.x;        // 0..31
    const int ls   = blockIdx.x;         // L split index
    const int nt   = blockIdx.y;         // n tile
    const int b    = blockIdx.z;

    const int row = lane & 15;           // A row / B col within tile
    const int kh  = lane >> 4;           // which K half this lane holds
    const int o1  = kh * 2;              // float4 offsets per ISA 16-bit A layout
    const int o2  = 4 + kh * 2;

    const int  n       = nt * 16 + row;
    const bool nValid  = (n < NN);
    const bool onesRow = (nt == NT - 1) && (row == 15);   // padded n==111 -> sum_t row

    // Clamped pointers: invalid lanes load in-bounds garbage that is zeroed
    // by the fma constants below, keeping the loop branchless / EXEC full.
    const float* aRow = pred_masks + (size_t)(b * NN + (nValid ? n : 0)) * LL;

    const int  m0      = row;                 // tile 0: m = 0..15 (all valid)
    const int  m1      = 16 + row;            // tile 1: m = 16..31
    const bool m1Valid = (m1 < MM);
    const bool onesCol = (m1 == MM);          // col 20 -> sum_p column
    const float* tRow0 = gt_masks + (size_t)(b * MM + m0) * LL;
    const float* tRow1 = gt_masks + (size_t)(b * MM + (m1Valid ? m1 : 0)) * LL;

    // Per-lane loop-invariant blend constants: val = raw*mul + add.
    const float aMul = nValid  ? 1.0f : 0.0f;
    const float aAdd = onesRow ? 1.0f : 0.0f;
    const float bMul = m1Valid ? 1.0f : 0.0f;
    const float bAdd = onesCol ? 1.0f : 0.0f;

    v8f c0 = {};   // accumulator, m-tile 0
    v8f c1 = {};   // accumulator, m-tile 1

    // Software pipeline: ping-pong tiles, 2 K-steps per pass.
    KTile tA, tB;
    int lbase = ls * LCHUNK;
    load_tiles(tA, aRow, tRow0, tRow1, lbase, o1, o2);

    #pragma unroll 1
    for (int it = 0; it < KITERS; it += 2) {
        const int l1 = lbase + 32;
        int       l2 = lbase + 64;            // lookahead; clamp final OOB step
        if (l2 > LL - 32) l2 = LL - 32;       // garbage tile, never processed

        // Prefetch ~4 K-steps ahead (speculative; dropped if OOB).
        __builtin_prefetch(aRow  + lbase + 128, 0, 3);
        __builtin_prefetch(tRow0 + lbase + 128, 0, 3);
        __builtin_prefetch(tRow1 + lbase + 128, 0, 3);

        load_tiles(tB, aRow, tRow0, tRow1, l1, o1, o2);      // step k+1 in flight
        process_tile(tA, aMul, aAdd, bMul, bAdd, c0, c1);    // consume step k

        load_tiles(tA, aRow, tRow0, tRow1, l2, o1, o2);      // step k+2 in flight
        process_tile(tB, aMul, aAdd, bMul, bAdd, c0, c1);    // consume step k+1

        lbase += 64;
    }

    // ---- Reduce L-splits: atomic f32 adds into ws[b][112][32] ----
    // C/D layout: lane holds D[v + 8*kh][lane&15] in VGPR v.
    float* base = ws + ((size_t)b * NP + nt * 16) * MP;
    #pragma unroll
    for (int v = 0; v < 8; ++v) {
        const int drow = v + 8 * kh;
        atomicAdd(base + drow * MP + row,      c0[v]);
        atomicAdd(base + drow * MP + 16 + row, c1[v]);
    }
}

// ---------------------------------------------------------------------------
// Epilogue: gather class cost, combine mask + dice terms.
// ---------------------------------------------------------------------------
__global__ void hung_finalize(const float* __restrict__ pred_logits,
                              const int*   __restrict__ gt_labels,
                              const float* __restrict__ ws,
                              float*       __restrict__ out) {
    const int idx = blockIdx.x * blockDim.x + threadIdx.x;
    if (idx >= BB * NN * MM) return;
    const int m = idx % MM;
    const int n = (idx / MM) % NN;
    const int b = idx / (MM * NN);

    const float* wb = ws + (size_t)b * NP * MP;
    const float Spt   = wb[n * MP + m];
    const float sum_p = wb[n * MP + MM];          // ones-column trick (col 20)
    const float sum_t = wb[(NP - 1) * MP + m];    // ones-row trick (row 111)

    const int   label = gt_labels[b * MM + m];
    const float cls   = -pred_logits[(size_t)(b * NN + n) * CC + label];

    const float invL  = 1.0f / (float)LL;
    const float s2    = 2.0f * Spt;
    const float cmask = (sum_p + sum_t - s2 - (float)LL) * invL;
    const float cdice = 1.0f - (s2 + 1.0f) / (sum_p + sum_t + 1.0f);

    out[idx] = cls + cmask + cdice;
}

// ---------------------------------------------------------------------------
extern "C" void kernel_launch(void* const* d_in, const int* in_sizes, int n_in,
                              void* d_out, int out_size, void* d_ws, size_t ws_size,
                              hipStream_t stream) {
    const float* pred_logits = (const float*)d_in[0];
    const float* pred_masks  = (const float*)d_in[1];
    const int*   gt_labels   = (const int*)  d_in[2];
    const float* gt_masks    = (const float*)d_in[3];
    float* out = (float*)d_out;
    float* ws  = (float*)d_ws;   // needs BB*NP*MP*4 = 57344 bytes

    const int zn = BB * NP * MP;
    hung_zero_ws<<<(zn + 255) / 256, 256, 0, stream>>>(ws);

    dim3 grid(LSPLIT, NT, BB);
    hung_cost_wmma<<<grid, 32, 0, stream>>>(pred_masks, gt_masks, ws);

    const int tot = BB * NN * MM;
    hung_finalize<<<(tot + 127) / 128, 128, 0, stream>>>(pred_logits, gt_labels, ws, out);
}
